// RPNHead_64553358459148
// MI455X (gfx1250) — matrix operand order repaired and checked by reading, not compiled, not run on hardware
//
#include <hip/hip_runtime.h>
#include <hip/hip_bf16.h>
#include <math.h>

typedef __attribute__((ext_vector_type(16))) _Float16 v16h;
typedef __attribute__((ext_vector_type(8)))  _Float16 v8h;
typedef __attribute__((ext_vector_type(8)))  float    v8f;

union V16 { v16h v; v8h h[2]; };

#define TOTAL_A 65472   // 3*(128^2+64^2+32^2+16^2+8^2)
#define KTOT    3960    // 1000+1000+1000+768+192
#define BBOX_CLIP 4.135166556742356f

// ---------------------------------------------------------------------------
// Pack conv weights [Cout=256][Cin=256][3][3] f32 -> f16 Wc[tap][co][ci]
// and head weights (cls 3x256 + bbox 12x256, padded to 16 rows) -> Wh[16][256]
// ---------------------------------------------------------------------------
__global__ __launch_bounds__(256) void pack_kernel(
    const float* __restrict__ cw, const float* __restrict__ clsw,
    const float* __restrict__ bbw, _Float16* __restrict__ Wc,
    _Float16* __restrict__ Wh)
{
    int i = blockIdx.x * 256 + threadIdx.x;
    const int NW = 9 * 256 * 256;
    if (i < NW) {
        int ci = i & 255;
        int co = (i >> 8) & 255;
        int t  = i >> 16;
        Wc[i] = (_Float16)cw[co * 2304 + ci * 9 + t];
    } else if (i < NW + 16 * 256) {
        int j = i - NW;
        int ci = j & 255, r = j >> 8;
        float v = 0.0f;
        if (r < 3)       v = clsw[r * 256 + ci];
        else if (r < 15) v = bbw[(r - 3) * 256 + ci];
        Wh[j] = (_Float16)v;
    }
}

// ---------------------------------------------------------------------------
// NCHW f32 -> NHWC f16 feature conversion (one level). Makes every patch row
// a contiguous 512B run so the conv kernel can stage it with a single
// wave-wide global_load_async_to_lds_b128.
// ---------------------------------------------------------------------------
__global__ __launch_bounds__(256) void convert_kernel(
    const float* __restrict__ in, _Float16* __restrict__ out, int H)
{
    const int W = H;
    size_t idx = (size_t)blockIdx.x * 256 + threadIdx.x;
    int ci = (int)(idx & 255);
    size_t rest = idx >> 8;
    int x = (int)(rest % W);
    int y = (int)((rest / W) % H);
    int n = (int)(rest / ((size_t)W * H));
    out[idx] = (_Float16)in[(((size_t)(n * 256 + ci)) * H + y) * W + x];
}

// ---------------------------------------------------------------------------
// Fused 3x3 conv (WMMA implicit GEMM) + ReLU + 1x1 head convs (WMMA) +
// anchor decode. Block handles (n, y, 16-wide x tile), 8 waves x 32 couts.
// Input patch staged via async global->LDS copies (ASYNCcnt path).
// ---------------------------------------------------------------------------
__global__ __launch_bounds__(256) void rpn_conv_kernel(
    const _Float16* __restrict__ feat16, const _Float16* __restrict__ Wc,
    const _Float16* __restrict__ Wh, const float* __restrict__ conv_b,
    const float* __restrict__ cls_b, const float* __restrict__ bbox_b,
    float* __restrict__ scores_all, float* __restrict__ boxes_all,
    int H, int stride, int sizep, int aoff)
{
    const int W = H;
    __shared__ alignas(16) _Float16 patch[3 * 18 * 264]; // input tile, f16
    __shared__ alignas(16) _Float16 ttile[16 * 264];     // relu(conv) tile
    __shared__ float htile[16 * 17];                     // head outputs

    const int tid  = threadIdx.x;
    const int lane = tid & 31, wv = tid >> 5;
    const int half = lane >> 4, l15 = lane & 15;
    const int x0 = blockIdx.x * 16, y = blockIdx.y, n = blockIdx.z;

    // ---- stage 3x18 rows of 256 f16 (512B each): async copy or zero halo ----
    // (segment validity is uniform across the wave -> EXEC stays all-ones)
    for (int s = wv; s < 54; s += 8) {
        int dy = s / 18, xx = s % 18;
        int gy = y + dy - 1, gx = x0 + xx - 1;
        _Float16* lrow = patch + (dy * 18 + xx) * 264;
        if (gy >= 0 && gy < H && gx >= 0 && gx < W) {
            const _Float16* g =
                feat16 + (((size_t)(n * H + gy)) * W + gx) * 256 + lane * 8;
            unsigned lds_b = (unsigned)(size_t)lrow + (unsigned)(lane * 16);
            unsigned long long ga = (unsigned long long)(size_t)g;
            asm volatile("global_load_async_to_lds_b128 %0, %1, off"
                         :: "v"(lds_b), "v"(ga) : "memory");
        } else {
            v8h z = {};
            *(v8h*)(lrow + lane * 8) = z;
        }
    }
    asm volatile("s_wait_asynccnt 0" ::: "memory");
    __syncthreads();

    // ---- implicit GEMM: M=16 positions, N=32 couts/wave, K=9*256 ----
    v8f acc0 = {}; v8f acc1 = {};
    const int co0 = wv * 32 + l15;
    const int co1 = co0 + 16;
    for (int t = 0; t < 9; ++t) {
        int dy = t / 3, dx = t % 3;
        const _Float16* arow = patch + (dy * 18 + l15 + dx) * 264;
        const _Float16* b0 = Wc + ((size_t)(t * 256 + co0) * 256) + half * 8;
        const _Float16* b1 = Wc + ((size_t)(t * 256 + co1) * 256) + half * 8;
        for (int kc = 0; kc < 8; ++kc) {
            int kb = kc * 32;
            V16 a, bA, bB;
            a.h[0]  = *(const v8h*)(arow + kb + half * 8);
            a.h[1]  = *(const v8h*)(arow + kb + 16 + half * 8);
            bA.h[0] = *(const v8h*)(b0 + kb);
            bA.h[1] = *(const v8h*)(b0 + kb + 16);
            bB.h[0] = *(const v8h*)(b1 + kb);
            bB.h[1] = *(const v8h*)(b1 + kb + 16);
            acc0 = __builtin_amdgcn_wmma_f32_16x16x32_f16(
                false, a.v, false, bA.v, (short)0, acc0, false, false);
            acc1 = __builtin_amdgcn_wmma_f32_16x16x32_f16(
                false, a.v, false, bB.v, (short)0, acc1, false, false);
        }
    }

    // ---- bias + ReLU -> f16 LDS tile [16 pos][256 co] ----
    {
        float cb0 = conv_b[co0], cb1 = conv_b[co1];
        #pragma unroll
        for (int v = 0; v < 8; ++v) {
            int m = v + half * 8;
            ttile[m * 264 + co0] = (_Float16)fmaxf(acc0[v] + cb0, 0.0f);
            ttile[m * 264 + co1] = (_Float16)fmaxf(acc1[v] + cb1, 0.0f);
        }
    }
    __syncthreads();

    // ---- fused 1x1 head convs: [16 pos x 16 heads] = ttile x Wh^T ----
    if (wv == 0) {
        v8f hacc = {};
        const _Float16* arow = ttile + l15 * 264;
        const _Float16* brow = Wh + l15 * 256 + half * 8;
        for (int kc = 0; kc < 8; ++kc) {
            int kb = kc * 32;
            V16 a, b;
            a.h[0] = *(const v8h*)(arow + kb + half * 8);
            a.h[1] = *(const v8h*)(arow + kb + 16 + half * 8);
            b.h[0] = *(const v8h*)(brow + kb);
            b.h[1] = *(const v8h*)(brow + kb + 16);
            hacc = __builtin_amdgcn_wmma_f32_16x16x32_f16(
                false, a.v, false, b.v, (short)0, hacc, false, false);
        }
        #pragma unroll
        for (int v = 0; v < 8; ++v) {
            int m = v + half * 8;
            htile[m * 17 + l15] = hacc[v];
        }
    }
    __syncthreads();

    // ---- anchor decode: 16 positions x 3 anchors ----
    if (tid < 48) {
        int m = tid / 3, a = tid % 3;
        int x = x0 + m;
        if (x < W) {
            float logit = htile[m * 17 + a] + cls_b[a];
            float d0 = htile[m * 17 + 3 + a * 4 + 0] + bbox_b[a * 4 + 0];
            float d1 = htile[m * 17 + 3 + a * 4 + 1] + bbox_b[a * 4 + 1];
            float d2 = fminf(htile[m * 17 + 3 + a * 4 + 2] + bbox_b[a * 4 + 2], BBOX_CLIP);
            float d3 = fminf(htile[m * 17 + 3 + a * 4 + 3] + bbox_b[a * 4 + 3], BBOX_CLIP);
            float ratio = (a == 0) ? 0.5f : (a == 1) ? 1.0f : 2.0f;
            float hr = sqrtf(ratio), wr = 1.0f / hr;
            float bx0 = roundf(-wr * sizep * 0.5f), by0 = roundf(-hr * sizep * 0.5f);
            float bx2 = roundf( wr * sizep * 0.5f), by2 = roundf( hr * sizep * 0.5f);
            float aw = bx2 - bx0, ah = by2 - by0;
            float acx = (float)(x * stride) + bx0 + 0.5f * aw;
            float acy = (float)(y * stride) + by0 + 0.5f * ah;
            float pcx = d0 * aw + acx, pcy = d1 * ah + acy;
            float pw = expf(d2) * aw, ph = expf(d3) * ah;
            float q0 = fminf(fmaxf(pcx - 0.5f * pw, 0.0f), 512.0f);
            float q1 = fminf(fmaxf(pcy - 0.5f * ph, 0.0f), 512.0f);
            float q2 = fminf(fmaxf(pcx + 0.5f * pw, 0.0f), 512.0f);
            float q3 = fminf(fmaxf(pcy + 0.5f * ph, 0.0f), 512.0f);
            int ai = aoff + (y * W + x) * 3 + a;
            scores_all[(size_t)n * TOTAL_A + ai] = logit;
            float* bp = boxes_all + ((size_t)n * TOTAL_A + ai) * 4;
            bp[0] = q0; bp[1] = q1; bp[2] = q2; bp[3] = q3;
        }
    }
}

// ---------------------------------------------------------------------------
// Per-(image, level) top-k via 256-bucket radix selection on ordered keys.
// ---------------------------------------------------------------------------
__device__ __forceinline__ unsigned fkey(float f) {
    unsigned u = __float_as_uint(f);
    return (u & 0x80000000u) ? ~u : (u | 0x80000000u);
}

__global__ __launch_bounds__(256) void topk_kernel(
    const float* __restrict__ scores_all, const float* __restrict__ boxes_all,
    float* __restrict__ cand_s, float* __restrict__ cand_b)
{
    __shared__ int hist[256];
    __shared__ int s_tb, s_cum, s_rem, s_hi, s_eq;
    const int Hs[5]   = {128, 64, 32, 16, 8};
    const int Aoff[5] = {0, 49152, 61440, 64512, 65280};
    const int Ks[5]   = {1000, 1000, 1000, 768, 192};
    const int Coff[5] = {0, 1000, 2000, 3000, 3768};

    int lvl = blockIdx.x, n = blockIdx.y, tid = threadIdx.x;
    int cnt = 3 * Hs[lvl] * Hs[lvl];
    int base = n * TOTAL_A + Aoff[lvl];
    int k = Ks[lvl];

    hist[tid] = 0;
    __syncthreads();
    for (int i = tid; i < cnt; i += 256)
        atomicAdd(&hist[fkey(scores_all[base + i]) >> 24], 1);
    __syncthreads();
    if (tid == 0) {
        int cum = 0, tb = 0, rem = k;
        for (int b = 255; b >= 0; --b) {
            int h = hist[b];
            if (cum + h >= k) { tb = b; rem = k - cum; break; }
            cum += h;
        }
        s_tb = tb; s_cum = cum; s_rem = rem; s_hi = 0; s_eq = 0;
    }
    __syncthreads();
    int tb = s_tb, cum = s_cum, rem = s_rem;
    for (int i = tid; i < cnt; i += 256) {
        float sc = scores_all[base + i];
        int b = (int)(fkey(sc) >> 24);
        int pos = -1;
        if (b > tb) pos = atomicAdd(&s_hi, 1);
        else if (b == tb) {
            int e = atomicAdd(&s_eq, 1);
            if (e < rem) pos = cum + e;
        }
        if (pos >= 0) {
            int o = n * KTOT + Coff[lvl] + pos;
            cand_s[o] = sc;
            const float* bp = boxes_all + (size_t)(base + i) * 4;
            float* cp = cand_b + (size_t)o * 4;
            cp[0] = bp[0]; cp[1] = bp[1]; cp[2] = bp[2]; cp[3] = bp[3];
        }
    }
}

// ---------------------------------------------------------------------------
// Greedy NMS: one block per image, 300 argmax/suppress iterations.
// Batched-NMS offset (513*lvl) applied on the fly; boxes streamed from L2.
// ---------------------------------------------------------------------------
__device__ __forceinline__ int lvl_of(int i) {
    return (i < 1000) ? 0 : (i < 2000) ? 1 : (i < 3000) ? 2 : (i < 3768) ? 3 : 4;
}

__global__ __launch_bounds__(256) void nms_kernel(
    const float* __restrict__ cand_s, const float* __restrict__ cand_b,
    float* __restrict__ outB, float* __restrict__ outS)
{
    __shared__ float msk[KTOT];
    __shared__ float rv[256];
    __shared__ int   ri[256];
    __shared__ float bb[8];

    int n = blockIdx.x, tid = threadIdx.x;

    for (int i = tid; i < KTOT; i += 256) {
        float lg = cand_s[n * KTOT + i];
        const float* cb = cand_b + (size_t)(n * KTOT + i) * 4;
        bool valid = (cb[2] - cb[0] >= 1e-3f) && (cb[3] - cb[1] >= 1e-3f);
        float prob = 1.0f / (1.0f + expf(-lg));
        msk[i] = valid ? prob : -1.0f;
    }
    __syncthreads();

    for (int step = 0; step < 300; ++step) {
        float lv = -1e30f; int li = 0;
        for (int i = tid; i < KTOT; i += 256) {
            float v = msk[i];
            if (v > lv) { lv = v; li = i; }
        }
        rv[tid] = lv; ri[tid] = li;
        __syncthreads();
        for (int s = 128; s > 0; s >>= 1) {
            if (tid < s) {
                float ov = rv[tid + s]; int oi = ri[tid + s];
                if (ov > rv[tid] || (ov == rv[tid] && oi < ri[tid])) {
                    rv[tid] = ov; ri[tid] = oi;
                }
            }
            __syncthreads();
        }
        if (tid == 0) {
            int best = ri[0]; float val = rv[0];
            const float* cb = cand_b + (size_t)(n * KTOT + best) * 4;
            float off = 513.0f * (float)lvl_of(best);
            bb[0] = cb[0] + off; bb[1] = cb[1] + off;
            bb[2] = cb[2] + off; bb[3] = cb[3] + off;
            bb[4] = val; ((int*)bb)[5] = best;
            float* ob = outB + (size_t)(n * 300 + step) * 4;
            if (val > -0.5f) {
                ob[0] = cb[0]; ob[1] = cb[1]; ob[2] = cb[2]; ob[3] = cb[3];
                outS[n * 300 + step] = val;
            } else {
                ob[0] = 0.f; ob[1] = 0.f; ob[2] = 0.f; ob[3] = 0.f;
                outS[n * 300 + step] = 0.f;
            }
        }
        __syncthreads();
        float v0 = bb[0], v1 = bb[1], v2 = bb[2], v3 = bb[3], val = bb[4];
        int best = ((int*)bb)[5];
        if (val > -0.5f) {
            float areaB = (v2 - v0) * (v3 - v1);
            for (int i = tid; i < KTOT; i += 256) {
                const float* cb = cand_b + (size_t)(n * KTOT + i) * 4;
                float off = 513.0f * (float)lvl_of(i);
                float q0 = cb[0] + off, q1 = cb[1] + off;
                float q2 = cb[2] + off, q3 = cb[3] + off;
                float iw = fmaxf(fminf(v2, q2) - fmaxf(v0, q0), 0.0f);
                float ih = fmaxf(fminf(v3, q3) - fmaxf(v1, q1), 0.0f);
                float inter = iw * ih;
                float areaQ = (q2 - q0) * (q3 - q1);
                float iou = inter / fmaxf(areaB + areaQ - inter, 1e-9f);
                if (iou > 0.7f) msk[i] = -1.0f;
            }
        }
        if (tid == 0) msk[best] = -1.0f;
        __syncthreads();
    }
}

// ---------------------------------------------------------------------------
extern "C" void kernel_launch(void* const* d_in, const int* in_sizes, int n_in,
                              void* d_out, int out_size, void* d_ws, size_t ws_size,
                              hipStream_t stream)
{
    (void)in_sizes; (void)n_in; (void)out_size; (void)ws_size;

    const float* feat[5];
    for (int i = 0; i < 5; ++i) feat[i] = (const float*)d_in[1 + i];
    const float* conv_w = (const float*)d_in[6];
    const float* conv_b = (const float*)d_in[7];
    const float* cls_w  = (const float*)d_in[8];
    const float* cls_b  = (const float*)d_in[9];
    const float* bbox_w = (const float*)d_in[10];
    const float* bbox_b = (const float*)d_in[11];

    const int Hs[5] = {128, 64, 32, 16, 8};
    const int St[5] = {4, 8, 16, 32, 64};
    const int Sz[5] = {16, 32, 64, 128, 256};
    const int Ao[5] = {0, 49152, 61440, 64512, 65280};

    // workspace carve (all offsets 16B aligned)
    char* w = (char*)d_ws;
    _Float16* Wc = (_Float16*)w;            w += (size_t)9 * 256 * 256 * 2;
    _Float16* Wh = (_Float16*)w;            w += (size_t)16 * 256 * 2;
    float* scores_all = (float*)w;          w += (size_t)4 * TOTAL_A * 4;
    float* boxes_all  = (float*)w;          w += (size_t)4 * TOTAL_A * 16;
    float* cand_s     = (float*)w;          w += (size_t)4 * KTOT * 4;
    float* cand_b     = (float*)w;          w += (size_t)4 * KTOT * 16;
    _Float16* feat16[5];
    for (int l = 0; l < 5; ++l) {
        feat16[l] = (_Float16*)w;
        w += (size_t)4 * Hs[l] * Hs[l] * 256 * 2;  // NHWC f16 per level
    }

    // 1) pack weights to f16
    {
        int total = 9 * 256 * 256 + 16 * 256;
        pack_kernel<<<(total + 255) / 256, 256, 0, stream>>>(
            conv_w, cls_w, bbox_w, Wc, Wh);
    }

    // 2) NCHW f32 -> NHWC f16 features (enables async-to-LDS staging)
    for (int l = 0; l < 5; ++l) {
        size_t total = (size_t)4 * Hs[l] * Hs[l] * 256;
        convert_kernel<<<(unsigned)(total / 256), 256, 0, stream>>>(
            feat[l], feat16[l], Hs[l]);
    }

    // 3) fused conv + heads + decode, per level
    for (int l = 0; l < 5; ++l) {
        dim3 grid((Hs[l] + 15) / 16, Hs[l], 4);
        rpn_conv_kernel<<<grid, 256, 0, stream>>>(
            feat16[l], Wc, Wh, conv_b, cls_b, bbox_b,
            scores_all, boxes_all, Hs[l], St[l], Sz[l], Ao[l]);
    }

    // 4) per-level top-k
    topk_kernel<<<dim3(5, 4), 256, 0, stream>>>(
        scores_all, boxes_all, cand_s, cand_b);

    // 5) NMS -> outputs (boxes [4,300,4] then scores [4,300])
    float* outB = (float*)d_out;
    float* outS = outB + (size_t)4 * 300 * 4;
    nms_kernel<<<4, 256, 0, stream>>>(cand_s, cand_b, outB, outS);
}
